// CausalSelfAttention_13709535609705
// MI455X (gfx1250) — matrix operand order repaired and checked by reading, not compiled, not run on hardware
//
#include <hip/hip_runtime.h>
#include <stdint.h>

// Causal self-attention for MI455X (gfx1250), bf16 WMMA pipeline with
// GLOBAL_LOAD_ASYNC_TO_LDS_B128 double-buffered staging (ASYNCcnt-tracked).
// ws (bytes): xb 16M | wabT 6M | wpbT 2M | q 16M | k 16M | v 16M | vT 16M | y 16M  (~104MB)

#define N_EMBD 1024
#define N_HEAD 16
#define HSIZE  64
#define TSEQ   2048
#define BATCH  4
#define M_ROWS (BATCH * TSEQ)   // 8192
#define N_QKV  (3 * N_EMBD)     // 3072

typedef __bf16 bf16_t;
typedef __attribute__((ext_vector_type(16))) __bf16 v16bf;
typedef __attribute__((ext_vector_type(8)))  float  v8f;

union Frag16 { unsigned u[8]; v16bf v; };

// ---- CDNA5 async global->LDS copy (no VGPR payload, tracked with ASYNCcnt) -------------
__device__ __forceinline__ void async_b128(void* lds_dst, const void* gsrc) {
  unsigned laddr = (unsigned)(size_t)(__attribute__((address_space(3))) char*)lds_dst;
  unsigned long long gaddr = (unsigned long long)gsrc;
  asm volatile("global_load_async_to_lds_b128 %0, %1, off"
               :: "v"(laddr), "v"(gaddr) : "memory");
}
__device__ __forceinline__ void wait_async0() {
#if __has_builtin(__builtin_amdgcn_s_wait_asynccnt)
  __builtin_amdgcn_s_wait_asynccnt(0);
#else
  asm volatile("s_wait_asynccnt 0x0" ::: "memory");
#endif
}

// A fragment: 16x32 bf16, row-major source, leading dim `ld` elements.
// lane = m + 16g; VGPR i holds K = 2i + 8g + 8*[i>=4] (pairs contiguous -> b32/b128 loads)
__device__ __forceinline__ v16bf load_a_frag(const bf16_t* src, int ld) {
  const int lane = threadIdx.x & 31;
  const int m = lane & 15, g = lane >> 4;
  Frag16 f;
#pragma unroll
  for (int i = 0; i < 8; ++i) {
    const int k0 = 2 * i + 8 * g + ((i >= 4) ? 8 : 0);
    f.u[i] = *reinterpret_cast<const unsigned*>(src + m * ld + k0);
  }
  return f.v;
}

// B fragment from [N][K] storage (B transposed): lane = n + 16g holds col n, K = 2i + 8g + 8*[i>=4],+1
__device__ __forceinline__ v16bf load_bt_frag(const bf16_t* src, int ld) {
  const int lane = threadIdx.x & 31;
  const int n = lane & 15, g = lane >> 4;
  Frag16 f;
#pragma unroll
  for (int i = 0; i < 8; ++i) {
    f.u[i] = *reinterpret_cast<const unsigned*>(src + n * ld + 2 * i + 8 * g + ((i >= 4) ? 8 : 0));
  }
  return f.v;
}

__device__ __forceinline__ v8f wmma_bf16(v16bf a, v16bf b, v8f c) {
  return __builtin_amdgcn_wmma_f32_16x16x32_bf16(false, a, false, b, (short)0, c, false, false);
}

__device__ __forceinline__ float hmax16(float v) {
  v = fmaxf(v, __shfl_xor(v, 1));
  v = fmaxf(v, __shfl_xor(v, 2));
  v = fmaxf(v, __shfl_xor(v, 4));
  v = fmaxf(v, __shfl_xor(v, 8));
  return v;
}
__device__ __forceinline__ float hsum16(float v) {
  v += __shfl_xor(v, 1);
  v += __shfl_xor(v, 2);
  v += __shfl_xor(v, 4);
  v += __shfl_xor(v, 8);
  return v;
}

// ---------------- Kernel 1: x f32 -> bf16 (straight copy) -------------------------------
__global__ void cvt_x_kernel(const float* __restrict__ x, bf16_t* __restrict__ xb) {
  const int total = M_ROWS * N_EMBD;
  const int stride = gridDim.x * blockDim.x;
  for (int i = blockIdx.x * blockDim.x + threadIdx.x; i < total; i += stride)
    xb[i] = (bf16_t)x[i];
}

// ---------------- Kernel 2: f32 [R][C] -> bf16 [C][R] (LDS-tiled, coalesced both sides) --
__global__ void __launch_bounds__(128) tcvt_kernel(const float* __restrict__ in,
                                                   bf16_t* __restrict__ outT, int R, int C) {
  __shared__ bf16_t tile[64 * 72];
  const int c0 = blockIdx.x * 64, r0 = blockIdx.y * 64;
#pragma unroll
  for (int i = 0; i < 32; ++i) {
    int lin = threadIdx.x + 128 * i;
    int tr = lin >> 6, tc = lin & 63;
    tile[tc * 72 + tr] = (bf16_t)in[(size_t)(r0 + tr) * C + c0 + tc];
  }
  __syncthreads();
#pragma unroll
  for (int i = 0; i < 32; ++i) {
    int lin = threadIdx.x + 128 * i;
    int tc = lin >> 6, tr = lin & 63;
    outT[(size_t)(c0 + tc) * R + r0 + tr] = tile[tc * 72 + tr];
  }
}

// ---------------- Kernel 3: qkv = x @ w_attn^T-layout, scatter q/k/v [B,H,T,hs] ----------
// 256 threads = 8 waves (2x4); 128x128 tile, BK=64, async double-buffered LDS pipeline.
__global__ void __launch_bounds__(256, 1) qkv_gemm_kernel(
    const bf16_t* __restrict__ xb, const bf16_t* __restrict__ wabT,
    bf16_t* __restrict__ qb, bf16_t* __restrict__ kb, bf16_t* __restrict__ vb) {
  __shared__ alignas(16) bf16_t ldsA[2][128 * 64];   // [row][k]
  __shared__ alignas(16) bf16_t ldsBT[2][128 * 64];  // [n][k]

  const int m0 = blockIdx.y * 128, n0 = blockIdx.x * 128;
  const int w = threadIdx.x >> 5, wm = w >> 2, wn = w & 3;
  const int lane = threadIdx.x & 31;
  const int n = lane & 15, g = lane >> 4;

  v8f acc[4][2];
  const v8f vzero = {0, 0, 0, 0, 0, 0, 0, 0};
#pragma unroll
  for (int i = 0; i < 4; ++i)
#pragma unroll
    for (int j = 0; j < 2; ++j) acc[i][j] = vzero;

  // async stage a 128x64 A-tile + 128x64 BT-tile into LDS buf (8 b128 per thread, no VGPRs)
  auto async_tile = [&](int k0, int buf) {
#pragma unroll
    for (int i = 0; i < 4; ++i) {
      int lin = threadIdx.x + 256 * i;           // 0..1023 -> 128 rows x 8 xuint4
      int row = lin >> 3, c4 = lin & 7;
      async_b128(ldsA[buf] + lin * 8,  xb   + (size_t)(m0 + row) * N_EMBD + k0 + c4 * 8);
      async_b128(ldsBT[buf] + lin * 8, wabT + (size_t)(n0 + row) * N_EMBD + k0 + c4 * 8);
    }
  };

  async_tile(0, 0);
  wait_async0();
  __syncthreads();

  const int NK = N_EMBD / 64;   // 16
  for (int kt = 0; kt < NK; ++kt) {
    if (kt + 1 < NK) async_tile((kt + 1) * 64, (kt + 1) & 1);  // overlaps compute below
    const bf16_t* As = ldsA[kt & 1];
    const bf16_t* Bs = ldsBT[kt & 1];
#pragma unroll
    for (int dc = 0; dc < 2; ++dc) {
      v16bf a[4], b[2];
#pragma unroll
      for (int i = 0; i < 4; ++i) a[i] = load_a_frag(As + (wm * 64 + i * 16) * 64 + dc * 32, 64);
#pragma unroll
      for (int j = 0; j < 2; ++j) b[j] = load_bt_frag(Bs + (wn * 32 + j * 16) * 64 + dc * 32, 64);
#pragma unroll
      for (int i = 0; i < 4; ++i)
#pragma unroll
        for (int j = 0; j < 2; ++j) acc[i][j] = wmma_bf16(a[i], b[j], acc[i][j]);
    }
    if (kt + 1 < NK) wait_async0();
    __syncthreads();
  }

#pragma unroll
  for (int i = 0; i < 4; ++i)
#pragma unroll
    for (int j = 0; j < 2; ++j)
#pragma unroll
      for (int r = 0; r < 8; ++r) {
        int row = m0 + wm * 64 + i * 16 + r + 8 * g;
        int col = n0 + wn * 32 + j * 16 + n;
        int which = col >> 10;
        int c2 = col & 1023;
        int h = c2 >> 6, d = c2 & 63;
        int bi = row >> 11, t = row & 2047;
        bf16_t* dst = (which == 0) ? qb : (which == 1) ? kb : vb;
        dst[(((bi * N_HEAD + h) * TSEQ) + t) * HSIZE + d] = (bf16_t)acc[i][j][r];
      }
}

// ---------------- Kernel 4: V [B,H,T,hs] -> V^T [B,H,hs,T] bf16 (LDS-tiled) --------------
__global__ void __launch_bounds__(128) transpose_v_kernel(const bf16_t* __restrict__ vb,
                                                          bf16_t* __restrict__ vtb) {
  __shared__ bf16_t tile[64 * 72];
  const int bh = blockIdx.x;
  const int t0 = blockIdx.y * 64;
  const bf16_t* src = vb + (size_t)bh * TSEQ * HSIZE;
  bf16_t* dst = vtb + (size_t)bh * TSEQ * HSIZE;
#pragma unroll
  for (int i = 0; i < 32; ++i) {
    int lin = threadIdx.x + 128 * i;
    int tr = lin >> 6, d = lin & 63;
    tile[d * 72 + tr] = src[(size_t)(t0 + tr) * HSIZE + d];
  }
  __syncthreads();
#pragma unroll
  for (int i = 0; i < 32; ++i) {
    int lin = threadIdx.x + 128 * i;
    int d = lin >> 6, tr = lin & 63;
    dst[(size_t)d * TSEQ + t0 + tr] = tile[d * 72 + tr];
  }
}

// ---------------- Kernel 5: flash-style causal attention, async double-buffered K/V^T ----
// 128 threads = 4 waves; wave w owns q rows [q0+16w, q0+16w+16).
__global__ void __launch_bounds__(128, 1) attn_kernel(
    const bf16_t* __restrict__ qb, const bf16_t* __restrict__ kb,
    const bf16_t* __restrict__ vtb, bf16_t* __restrict__ yb) {
  __shared__ alignas(16) bf16_t ldsQ[64 * 64];
  __shared__ alignas(16) bf16_t ldsK[2][32 * 64];    // [key][d] == [N][K] for QK^T
  __shared__ alignas(16) bf16_t ldsVT[2][64 * 32];   // [d][key] == [N][K] for P@V
  __shared__ alignas(16) bf16_t ldsP[4 * 16 * 32];

  const int bh = blockIdx.x;
  const int bi = bh >> 4, h = bh & 15;
  const int q0 = blockIdx.y * 64;
  const size_t base = (size_t)bh * TSEQ * HSIZE;

  {  // stage Q tile (64x64): 512 uint4, 4 per thread
    const uint4* qs = (const uint4*)(qb + base + (size_t)q0 * HSIZE);
    uint4* qd = (uint4*)ldsQ;
#pragma unroll
    for (int i = 0; i < 4; ++i) qd[threadIdx.x + 128 * i] = qs[threadIdx.x + 128 * i];
  }

  const int w = threadIdx.x >> 5;
  const int lane = threadIdx.x & 31;
  const int n = lane & 15, g = lane >> 4;
  const float NEG = -1.0e30f;
  const float SCALE = 0.125f;   // 1/sqrt(64)

  v8f accO[4];
  const v8f vzero = {0, 0, 0, 0, 0, 0, 0, 0};
#pragma unroll
  for (int jd = 0; jd < 4; ++jd) accO[jd] = vzero;
  float mrow[8], lrow[8];
#pragma unroll
  for (int r = 0; r < 8; ++r) { mrow[r] = NEG; lrow[r] = 0.f; }

  // async stage one 32x64 K-tile + 64x32 V^T-tile (4 b128 per thread)
  auto async_kv = [&](int kt, int buf) {
    const int kbase = kt * 32;
#pragma unroll
    for (int i = 0; i < 2; ++i) {
      int lin = threadIdx.x + 128 * i;  // 0..255
      async_b128(ldsK[buf] + lin * 8,
                 kb + base + (size_t)(kbase + (lin >> 3)) * HSIZE + (lin & 7) * 8);
      async_b128(ldsVT[buf] + lin * 8,
                 vtb + base + (size_t)(lin >> 2) * TSEQ + kbase + (lin & 3) * 8);
    }
  };

  bf16_t* ps = ldsP + w * (16 * 32);
  const int nkt = (q0 >> 5) + 2;   // key tiles of 32 covering keys [0, q0+64)

  async_kv(0, 0);
  wait_async0();
  __syncthreads();

  for (int kt = 0; kt < nkt; ++kt) {
    const int kbase = kt * 32;
    if (kt + 1 < nkt) async_kv(kt + 1, (kt + 1) & 1);  // overlaps compute below
    const bf16_t* Ks = ldsK[kt & 1];
    const bf16_t* Vs = ldsVT[kt & 1];

    // S = Q K^T  (16q x 32keys per wave)
    v8f s0 = vzero, s1 = vzero;
#pragma unroll
    for (int dc = 0; dc < 2; ++dc) {
      v16bf aq  = load_a_frag(ldsQ + (w * 16) * 64 + dc * 32, 64);
      v16bf bk0 = load_bt_frag(Ks + 0 * 64 + dc * 32, 64);
      v16bf bk1 = load_bt_frag(Ks + 16 * 64 + dc * 32, 64);
      s0 = wmma_bf16(aq, bk0, s0);
      s1 = wmma_bf16(aq, bk1, s1);
    }

    // causal mask + online softmax (row M = r + 8g striped across 16 lanes per half-wave)
    float alpha_r[8];
#pragma unroll
    for (int r = 0; r < 8; ++r) {
      const int qab = q0 + w * 16 + r + 8 * g;
      const bool msk0 = (kbase + n) > qab;
      const bool msk1 = (kbase + 16 + n) > qab;
      float v0 = msk0 ? NEG : s0[r] * SCALE;
      float v1 = msk1 ? NEG : s1[r] * SCALE;
      float t = hmax16(fmaxf(v0, v1));
      float mn = fmaxf(mrow[r], t);
      float al = __expf(mrow[r] - mn);
      float p0 = msk0 ? 0.f : __expf(v0 - mn);
      float p1 = msk1 ? 0.f : __expf(v1 - mn);
      lrow[r] = lrow[r] * al + hsum16(p0 + p1);
      mrow[r] = mn;
      alpha_r[r] = al;
      ps[(r + 8 * g) * 32 + n]      = (bf16_t)p0;
      ps[(r + 8 * g) * 32 + 16 + n] = (bf16_t)p1;
    }
#pragma unroll
    for (int jd = 0; jd < 4; ++jd)
#pragma unroll
      for (int r = 0; r < 8; ++r) accO[jd][r] *= alpha_r[r];

    // O += P @ V  (A: P 16x32; B: V^T [d][key] = [N][K], ld=32 -> b128 LDS loads)
    v16bf ap = load_a_frag(ps, 32);   // same-wave LDS RAW: DS in-order
#pragma unroll
    for (int jd = 0; jd < 4; ++jd) {
      v16bf bv = load_bt_frag(Vs + (jd * 16) * 32, 32);
      accO[jd] = wmma_bf16(ap, bv, accO[jd]);
    }

    if (kt + 1 < nkt) wait_async0();
    __syncthreads();
  }

  // normalize and write y [B,T,C] bf16
#pragma unroll
  for (int jd = 0; jd < 4; ++jd)
#pragma unroll
    for (int r = 0; r < 8; ++r) {
      float li = lrow[r];
      float o = (li > 0.f) ? accO[jd][r] / li : 0.f;
      int t = q0 + w * 16 + r + 8 * g;
      int c = h * HSIZE + jd * 16 + n;
      yb[((size_t)bi * TSEQ + t) * N_EMBD + c] = (bf16_t)o;
    }
}

// ---------------- Kernel 6: out = y @ w_proj (f32 output), same async pipelined GEMM -----
__global__ void __launch_bounds__(256, 1) proj_gemm_kernel(
    const bf16_t* __restrict__ yb, const bf16_t* __restrict__ wpbT, float* __restrict__ out) {
  __shared__ alignas(16) bf16_t ldsA[2][128 * 64];
  __shared__ alignas(16) bf16_t ldsBT[2][128 * 64];

  const int m0 = blockIdx.y * 128, n0 = blockIdx.x * 128;
  const int w = threadIdx.x >> 5, wm = w >> 2, wn = w & 3;
  const int lane = threadIdx.x & 31;
  const int n = lane & 15, g = lane >> 4;

  v8f acc[4][2];
  const v8f vzero = {0, 0, 0, 0, 0, 0, 0, 0};
#pragma unroll
  for (int i = 0; i < 4; ++i)
#pragma unroll
    for (int j = 0; j < 2; ++j) acc[i][j] = vzero;

  auto async_tile = [&](int k0, int buf) {
#pragma unroll
    for (int i = 0; i < 4; ++i) {
      int lin = threadIdx.x + 256 * i;
      int row = lin >> 3, c4 = lin & 7;
      async_b128(ldsA[buf] + lin * 8,  yb   + (size_t)(m0 + row) * N_EMBD + k0 + c4 * 8);
      async_b128(ldsBT[buf] + lin * 8, wpbT + (size_t)(n0 + row) * N_EMBD + k0 + c4 * 8);
    }
  };

  async_tile(0, 0);
  wait_async0();
  __syncthreads();

  const int NK = N_EMBD / 64;
  for (int kt = 0; kt < NK; ++kt) {
    if (kt + 1 < NK) async_tile((kt + 1) * 64, (kt + 1) & 1);
    const bf16_t* As = ldsA[kt & 1];
    const bf16_t* Bs = ldsBT[kt & 1];
#pragma unroll
    for (int dc = 0; dc < 2; ++dc) {
      v16bf a[4], b[2];
#pragma unroll
      for (int i = 0; i < 4; ++i) a[i] = load_a_frag(As + (wm * 64 + i * 16) * 64 + dc * 32, 64);
#pragma unroll
      for (int j = 0; j < 2; ++j) b[j] = load_bt_frag(Bs + (wn * 32 + j * 16) * 64 + dc * 32, 64);
#pragma unroll
      for (int i = 0; i < 4; ++i)
#pragma unroll
        for (int j = 0; j < 2; ++j) acc[i][j] = wmma_bf16(a[i], b[j], acc[i][j]);
    }
    if (kt + 1 < NK) wait_async0();
    __syncthreads();
  }

#pragma unroll
  for (int i = 0; i < 4; ++i)
#pragma unroll
    for (int j = 0; j < 2; ++j)
#pragma unroll
      for (int r = 0; r < 8; ++r) {
        int row = m0 + wm * 64 + i * 16 + r + 8 * g;
        int col = n0 + wn * 32 + j * 16 + n;
        out[(size_t)row * N_EMBD + col] = acc[i][j][r];
      }
}

// ---------------------------------------------------------------------------------------
extern "C" void kernel_launch(void* const* d_in, const int* in_sizes, int n_in,
                              void* d_out, int out_size, void* d_ws, size_t ws_size,
                              hipStream_t stream) {
  (void)in_sizes; (void)n_in; (void)out_size; (void)ws_size;
  const float* x      = (const float*)d_in[0];
  const float* w_attn = (const float*)d_in[1];
  const float* w_proj = (const float*)d_in[2];
  float* out = (float*)d_out;

  char* p = (char*)d_ws;
  bf16_t* xb   = (bf16_t*)p; p += (size_t)M_ROWS * N_EMBD * 2;   // 16 MB
  bf16_t* wabT = (bf16_t*)p; p += (size_t)N_QKV * N_EMBD * 2;    // 6 MB  [3072][1024]
  bf16_t* wpbT = (bf16_t*)p; p += (size_t)N_EMBD * N_EMBD * 2;   // 2 MB  [1024][1024]
  bf16_t* qb   = (bf16_t*)p; p += (size_t)M_ROWS * N_EMBD * 2;   // 16 MB
  bf16_t* kb   = (bf16_t*)p; p += (size_t)M_ROWS * N_EMBD * 2;   // 16 MB
  bf16_t* vb   = (bf16_t*)p; p += (size_t)M_ROWS * N_EMBD * 2;   // 16 MB
  bf16_t* vtb  = (bf16_t*)p; p += (size_t)M_ROWS * N_EMBD * 2;   // 16 MB
  bf16_t* yb   = (bf16_t*)p; p += (size_t)M_ROWS * N_EMBD * 2;   // 16 MB

  cvt_x_kernel<<<dim3(4096), dim3(256), 0, stream>>>(x, xb);
  tcvt_kernel<<<dim3(N_QKV / 64, N_EMBD / 64), dim3(128), 0, stream>>>(w_attn, wabT, N_EMBD, N_QKV);
  tcvt_kernel<<<dim3(N_EMBD / 64, N_EMBD / 64), dim3(128), 0, stream>>>(w_proj, wpbT, N_EMBD, N_EMBD);
  qkv_gemm_kernel<<<dim3(N_QKV / 128, M_ROWS / 128), dim3(256), 0, stream>>>(xb, wabT, qb, kb, vb);
  transpose_v_kernel<<<dim3(BATCH * N_HEAD, TSEQ / 64), dim3(128), 0, stream>>>(vb, vtb);
  attn_kernel<<<dim3(BATCH * N_HEAD, TSEQ / 64), dim3(128), 0, stream>>>(qb, kb, vtb, yb);
  proj_gemm_kernel<<<dim3(N_EMBD / 128, M_ROWS / 128), dim3(256), 0, stream>>>(yb, wpbT, out);
}
